// HTRBiMambaHybrid_27376121544956
// MI455X (gfx1250) — matrix operand order repaired and checked by reading, not compiled, not run on hardware
//
#include <hip/hip_runtime.h>
#include <hip/hip_bf16.h>
#include <math.h>

// ---------------------------------------------------------------------------
// HTR BiMamba hybrid for gfx1250 (MI455X).  All GEMMs use
// v_wmma_f32_16x16x32_f16 (wave32 WMMA) with 64x32 register tiling per wave.
// Attention is flash-style (no materialized TxT scores), templated on head
// dim, with row-major softmax through a single LDS transpose.  Selective scan
// is a register-resident 16-state recurrence, one thread per (batch, channel).
// ---------------------------------------------------------------------------

typedef __attribute__((ext_vector_type(16))) _Float16 v16h;
typedef __attribute__((ext_vector_type(8)))  float    v8f;

#define NB    16
#define NT    1024
#define DMODEL 128
#define DINNER 256
#define NTOK  (NB*NT)        // 16384

#define GOP_STORE 0
#define GOP_ACCUM 1
#define GOP_GELU  2

// ---------------------------------------------------------------------------
// Generic GEMM:  Out[M,N] (+)= X[M,K] @ W[N,K]^T (+ bias) (with epilogue op)
// One wave computes a 64x32 tile (4 x 2 WMMA 16x16 tiles), K templated so all
// load offsets are immediates.  M % 64 == 0 required; N arbitrary (guarded).
// ---------------------------------------------------------------------------
template<int K>
__global__ void __launch_bounds__(128)
gemm_wmma_kernel(const float* __restrict__ X, const float* __restrict__ W,
                 const float* __restrict__ bias, float* __restrict__ Out,
                 int M, int N, int ntilesN, int totalWaves, int mode)
{
    constexpr int TM = 4, TN = 2;
    int wid = blockIdx.x * 4 + (threadIdx.x >> 5);
    if (wid >= totalWaves) return;
    int lane = threadIdx.x & 31;
    int lm = lane & 15;          // col (C layout) / row m (A layout)
    int lh = lane >> 4;          // lane half

    int tm = wid / ntilesN;
    int tn = wid % ntilesN;
    int m0 = tm * (16 * TM);
    int n0 = tn * (16 * TN);

    v8f acc[TM][TN] = {};

    const float* abase = X + (size_t)(m0 + lm) * K;      // per-lane A row
    const float* wbase = W + (size_t)(n0 + lm) * K;      // per-lane B row (n=lm)
    // wave-uniform predicate -> scalar branch
    int fullS = __builtin_amdgcn_readfirstlane((n0 + 16 * TN) <= N ? 1 : 0);

    for (int k0 = 0; k0 < K; k0 += 32) {
        v16h a[TM];
        #pragma unroll
        for (int i = 0; i < TM; ++i) {
            #pragma unroll
            for (int j = 0; j < 8; ++j) {
                int kidx = k0 + 16 * (j >> 2) + 8 * lh + 2 * (j & 3);
                a[i][2*j]   = (_Float16)abase[i * 16 * K + kidx];
                a[i][2*j+1] = (_Float16)abase[i * 16 * K + kidx + 1];
            }
        }
        v16h bf[TN];
        if (fullS) {
            #pragma unroll
            for (int t = 0; t < TN; ++t) {
                #pragma unroll
                for (int j = 0; j < 8; ++j) {
                    int kidx = k0 + 16 * (j >> 2) + 8 * lh + 2 * (j & 3);
                    bf[t][2*j]   = (_Float16)wbase[t * 16 * K + kidx];
                    bf[t][2*j+1] = (_Float16)wbase[t * 16 * K + kidx + 1];
                }
            }
        } else {
            #pragma unroll
            for (int t = 0; t < TN; ++t) {
                #pragma unroll
                for (int j = 0; j < 8; ++j) {
                    int na = n0 + t * 16 + lm;
                    int kidx = k0 + 16 * (j >> 2) + 8 * lh + 2 * (j & 3);
                    float va = W[(size_t)min(na, N - 1) * K + kidx];
                    float vb = W[(size_t)min(na, N - 1) * K + kidx + 1];
                    bf[t][2*j]   = (na < N) ? (_Float16)va : (_Float16)0.f;
                    bf[t][2*j+1] = (na < N) ? (_Float16)vb : (_Float16)0.f;
                }
            }
        }
        #pragma unroll
        for (int i = 0; i < TM; ++i)
            #pragma unroll
            for (int t = 0; t < TN; ++t)
                acc[i][t] = __builtin_amdgcn_wmma_f32_16x16x32_f16(
                                false, a[i], false, bf[t], (short)0, acc[i][t], false, false);
    }

    #pragma unroll
    for (int i = 0; i < TM; ++i) {
        int rowbase = m0 + i * 16 + 8 * lh;
        #pragma unroll
        for (int t = 0; t < TN; ++t) {
            int n = n0 + t * 16 + lm;
            if (n < N) {
                float bv = bias ? bias[n] : 0.f;
                #pragma unroll
                for (int r = 0; r < 8; ++r) {
                    float v = acc[i][t][r] + bv;
                    size_t idx = (size_t)(rowbase + r) * N + n;
                    if (mode == GOP_ACCUM)      Out[idx] += v;
                    else if (mode == GOP_GELU)  Out[idx] = 0.5f * v * (1.f + erff(v * 0.70710678118f));
                    else                        Out[idx] = v;
                }
            }
        }
    }
}

// ---------------------------------------------------------------------------
// LayerNorm over D=128, one wave per token.
// ---------------------------------------------------------------------------
__global__ void ln128_kernel(const float* __restrict__ in, const float* __restrict__ w,
                             const float* __restrict__ b, float* __restrict__ out,
                             int ntok, float eps)
{
    int wid  = (blockIdx.x * blockDim.x + threadIdx.x) >> 5;
    int lane = threadIdx.x & 31;
    if (wid >= ntok) return;
    const float* p = in + (size_t)wid * 128;
    float v0 = p[lane], v1 = p[lane+32], v2 = p[lane+64], v3 = p[lane+96];
    float s = v0 + v1 + v2 + v3;
    #pragma unroll
    for (int m = 16; m >= 1; m >>= 1) s += __shfl_xor(s, m, 32);
    float mean = s * (1.f / 128.f);
    float d0 = v0-mean, d1 = v1-mean, d2 = v2-mean, d3 = v3-mean;
    float vs = d0*d0 + d1*d1 + d2*d2 + d3*d3;
    #pragma unroll
    for (int m = 16; m >= 1; m >>= 1) vs += __shfl_xor(vs, m, 32);
    float r = rsqrtf(vs * (1.f / 128.f) + eps);
    float* q = out + (size_t)wid * 128;
    q[lane]    = d0 * r * w[lane]    + b[lane];
    q[lane+32] = d1 * r * w[lane+32] + b[lane+32];
    q[lane+64] = d2 * r * w[lane+64] + b[lane+64];
    q[lane+96] = d3 * r * w[lane+96] + b[lane+96];
}

// ---------------------------------------------------------------------------
// Stem helpers: mean over Hf=8, then im2col for the stride-4 K=4 conv.
// ---------------------------------------------------------------------------
__global__ void mean_h_kernel(const float* __restrict__ x, float* __restrict__ feats, int total)
{
    int i = blockIdx.x * blockDim.x + threadIdx.x;
    if (i >= total) return;
    int w  = i & 4095;
    int bc = i >> 12;
    const float* p = x + (size_t)bc * 8 * 4096 + w;
    float s = 0.f;
    #pragma unroll
    for (int h = 0; h < 8; ++h) s += p[(size_t)h * 4096];
    feats[i] = s * 0.125f;
}

__global__ void im2col_kernel(const float* __restrict__ feats, float* __restrict__ X, int total)
{
    int i = blockIdx.x * blockDim.x + threadIdx.x;
    if (i >= total) return;
    int q   = i & 511;
    int tok = i >> 9;
    int b = tok >> 10, t = tok & 1023;
    int ci = q >> 2,  k = q & 3;
    X[i] = feats[((size_t)(b * 128 + ci)) * 4096 + 4 * t + k];
}

// ---------------------------------------------------------------------------
// Depthwise causal conv (K=4) + SiLU.  dir=0: y[t]=sum w[j]*x[t-3+j]
// dir=1 (reverse-time mamba): y[t]=sum w[j]*x[t+3-j]
// ---------------------------------------------------------------------------
__global__ void conv_silu_kernel(const float* __restrict__ xz, const float* __restrict__ cw,
                                 const float* __restrict__ cb, float* __restrict__ xc,
                                 int dir, int total)
{
    int i = blockIdx.x * blockDim.x + threadIdx.x;
    if (i >= total) return;
    int d = i & 255;
    int tok = i >> 8;
    int b = tok >> 10, t = tok & 1023;
    float acc = cb[d];
    #pragma unroll
    for (int j = 0; j < 4; ++j) {
        int tt = dir ? (t + 3 - j) : (t - 3 + j);
        if (tt >= 0 && tt < NT)
            acc = fmaf(cw[d * 4 + j], xz[((size_t)(b << 10) + tt) * 512 + d], acc);
    }
    xc[i] = acc / (1.f + __expf(-acc));   // silu
}

// ---------------------------------------------------------------------------
// delta = softplus(dbc[:, :8] @ dt_w^T + dt_b)     (K=8, plain VALU kernel)
// ---------------------------------------------------------------------------
__global__ void dt_softplus_kernel(const float* __restrict__ dbc, const float* __restrict__ dtw,
                                   const float* __restrict__ dtb, float* __restrict__ delta,
                                   int total)
{
    int i = blockIdx.x * blockDim.x + threadIdx.x;
    if (i >= total) return;
    int d = i & 255;
    int tok = i >> 8;
    const float* r = dbc + (size_t)tok * 40;
    float acc = dtb[d];
    #pragma unroll
    for (int j = 0; j < 8; ++j) acc = fmaf(r[j], dtw[d * 8 + j], acc);
    delta[i] = (acc > 20.f) ? acc : log1pf(__expf(acc));
}

// ---------------------------------------------------------------------------
// Selective scan.  One block per batch (256 threads = D_INNER channels).
// 16 recurrent states per thread in registers; B/C broadcast via LDS.
// Fused D skip-term and z (SiLU) gating.
// ---------------------------------------------------------------------------
__global__ void __launch_bounds__(256)
scan_kernel(const float* __restrict__ delta, const float* __restrict__ dbc,
            const float* __restrict__ xc, const float* __restrict__ xz,
            const float* __restrict__ Alog, const float* __restrict__ Dp,
            float* __restrict__ y, int dir)
{
    __shared__ float sB[16], sC[16];
    int b = blockIdx.x;
    int d = threadIdx.x;
    float A[16], h[16];
    #pragma unroll
    for (int s = 0; s < 16; ++s) { A[s] = -__expf(Alog[d * 16 + s]); h[s] = 0.f; }
    float Dv = Dp[d];

    for (int step = 0; step < NT; ++step) {
        int t = dir ? (NT - 1 - step) : step;
        size_t tok = (size_t)b * NT + t;
        __syncthreads();
        if (d < 16)       sB[d]      = dbc[tok * 40 + 8  + d];
        else if (d < 32)  sC[d - 16] = dbc[tok * 40 + 24 + (d - 16)];
        __syncthreads();
        float de = delta[tok * 256 + d];
        float xv = xc[tok * 256 + d];
        float dx = de * xv;
        float accy = 0.f;
        #pragma unroll
        for (int s = 0; s < 16; ++s) {
            h[s] = h[s] * __expf(de * A[s]) + dx * sB[s];
            accy = fmaf(h[s], sC[s], accy);
        }
        float yv = accy + xv * Dv;
        float z  = xz[tok * 512 + 256 + d];
        yv *= z / (1.f + __expf(-z));     // * silu(z)
        y[tok * 256 + d] = yv;
    }
}

// ---------------------------------------------------------------------------
// Flash attention, one wave per (b, head, 16-query tile).  Templated on head
// dim HD (32 or 64) so all fragment arrays have constant indices.  WMMA for
// QK^T and P.V.  Softmax is row-major: raw S goes through one f32 LDS
// transpose; row max/sum are lane-local + one xor-16 shuffle; P A-fragment is
// built directly in registers; C-layout rescales use 8 lane broadcasts.
// ---------------------------------------------------------------------------
template<int HD>
__global__ void __launch_bounds__(128)
attn_flash_kernel(const float* __restrict__ qkv, float* __restrict__ out, int totalWaves)
{
    constexpr int NH  = 128 / HD;
    constexpr int NKQ = HD / 32;   // Q k-frags
    constexpr int NO  = HD / 16;   // O n-frags
    __shared__ float lsS[4][16][32];

    int wid = blockIdx.x * 4 + (threadIdx.x >> 5);
    if (wid >= totalWaves) return;
    int wv   = threadIdx.x >> 5;
    int lane = threadIdx.x & 31;
    int lm = lane & 15, lh = lane >> 4;

    int qt = wid & 63;
    int h  = (wid >> 6) % NH;
    int b  = wid / (64 * NH);
    float qscale = rsqrtf((float)HD);

    size_t tq0 = (size_t)(b * NT + qt * 16);

    v16h qa[NKQ];
    #pragma unroll
    for (int kk = 0; kk < NKQ; ++kk) {
        const float* qrow = qkv + (tq0 + lm) * 384 + h * HD + kk * 32;
        #pragma unroll
        for (int j = 0; j < 8; ++j) {
            int kidx = 16 * (j >> 2) + 8 * lh + 2 * (j & 3);
            qa[kk][2*j]   = (_Float16)(qrow[kidx]     * qscale);
            qa[kk][2*j+1] = (_Float16)(qrow[kidx + 1] * qscale);
        }
    }

    v8f o[NO] = {};
    float mrowA = -3.0e38f, lrowA = 0.f;   // row stats, row = lane & 15

    const float* kbase = qkv + (size_t)b * NT * 384 + 128 + h * HD;
    const float* vbase = qkv + (size_t)b * NT * 384 + 256 + h * HD;

    for (int kt = 0; kt < NT / 32; ++kt) {
        int key0 = kt * 32;
        v8f s0 = {}, s1 = {};
        #pragma unroll
        for (int kk = 0; kk < NKQ; ++kk) {
            // B for K^T: lane%16 = key (n), elements -> feature k (table order)
            const float* kl = kbase + (size_t)(key0 + lm) * 384 + kk * 32;
            v16h bk0, bk1;
            #pragma unroll
            for (int j = 0; j < 8; ++j) {
                int kidx = 16 * (j >> 2) + 8 * lh + 2 * (j & 3);
                bk0[2*j]   = (_Float16)kl[kidx];
                bk0[2*j+1] = (_Float16)kl[kidx + 1];
                bk1[2*j]   = (_Float16)kl[16 * 384 + kidx];
                bk1[2*j+1] = (_Float16)kl[16 * 384 + kidx + 1];
            }
            s0 = __builtin_amdgcn_wmma_f32_16x16x32_f16(false, qa[kk], false, bk0,
                                                        (short)0, s0, false, false);
            s1 = __builtin_amdgcn_wmma_f32_16x16x32_f16(false, qa[kk], false, bk1,
                                                        (short)0, s1, false, false);
        }
        // S: C layout -> LDS (f32)
        #pragma unroll
        for (int r = 0; r < 8; ++r) {
            lsS[wv][r + 8 * lh][lm]      = s0[r];
            lsS[wv][r + 8 * lh][16 + lm] = s1[r];
        }
        // row-major reload: this lane owns row lm, k in {8lh..8lh+7, 16+8lh..}
        float sv[16];
        #pragma unroll
        for (int v = 0; v < 8; ++v) {
            sv[v]     = lsS[wv][lm][8 * lh + v];
            sv[8 + v] = lsS[wv][lm][16 + 8 * lh + v];
        }
        float tmax = sv[0];
        #pragma unroll
        for (int v = 1; v < 16; ++v) tmax = fmaxf(tmax, sv[v]);
        tmax = fmaxf(tmax, __shfl_xor(tmax, 16, 32));
        float mn   = fmaxf(mrowA, tmax);
        float sclA = __expf(mrowA - mn);
        float p[16], rs = 0.f;
        #pragma unroll
        for (int v = 0; v < 16; ++v) { p[v] = __expf(sv[v] - mn); rs += p[v]; }
        rs += __shfl_xor(rs, 16, 32);
        lrowA = lrowA * sclA + rs;
        mrowA = mn;
        // P A-fragment directly from registers (k order matches lane's sv set)
        v16h pa;
        #pragma unroll
        for (int j = 0; j < 8; ++j) {
            int idx = ((j >> 2) << 3) + 2 * (j & 3);
            pa[2*j]   = (_Float16)p[idx];
            pa[2*j+1] = (_Float16)p[idx + 1];
        }
        // rescale O: broadcast row scales to C-layout rows r + 8*lh
        #pragma unroll
        for (int r = 0; r < 8; ++r) {
            float sc = __shfl(sclA, r + 8 * lh, 32);
            #pragma unroll
            for (int jn = 0; jn < NO; ++jn) o[jn][r] *= sc;
        }
        // O += P @ V   (B: lane%16 = feature col (n), elements -> key k (table order))
        const float* vrow = vbase + (size_t)key0 * 384 + lm;
        #pragma unroll
        for (int jn = 0; jn < NO; ++jn) {
            v16h bv;
            #pragma unroll
            for (int j = 0; j < 8; ++j) {
                int kidx = 16 * (j >> 2) + 8 * lh + 2 * (j & 3);
                bv[2*j]   = (_Float16)vrow[(size_t)kidx * 384 + jn * 16];
                bv[2*j+1] = (_Float16)vrow[(size_t)(kidx + 1) * 384 + jn * 16];
            }
            o[jn] = __builtin_amdgcn_wmma_f32_16x16x32_f16(false, pa, false, bv,
                                                           (short)0, o[jn], false, false);
        }
    }

    #pragma unroll
    for (int r = 0; r < 8; ++r) {
        float inv = 1.f / __shfl(lrowA, r + 8 * lh, 32);
        size_t row = tq0 + r + 8 * lh;
        #pragma unroll
        for (int jn = 0; jn < NO; ++jn)
            out[row * 128 + h * HD + jn * 16 + lm] = o[jn][r] * inv;
    }
}

// ---------------------------------------------------------------------------
// Host-side orchestration
// ---------------------------------------------------------------------------
static inline int cdiv(int a, int b) { return (a + b - 1) / b; }

static void launch_gemm(const float* X, const float* W, const float* bias, float* Out,
                        int M, int N, int K, int mode, hipStream_t s)
{
    int ntilesN = cdiv(N, 32);          // TN = 2
    int waves   = (M / 64) * ntilesN;   // TM = 4
    int blocks  = cdiv(waves, 4);
    switch (K) {
    case 128:
        gemm_wmma_kernel<128><<<blocks, 128, 0, s>>>(X, W, bias, Out, M, N, ntilesN, waves, mode);
        break;
    case 256:
        gemm_wmma_kernel<256><<<blocks, 128, 0, s>>>(X, W, bias, Out, M, N, ntilesN, waves, mode);
        break;
    case 512:
        gemm_wmma_kernel<512><<<blocks, 128, 0, s>>>(X, W, bias, Out, M, N, ntilesN, waves, mode);
        break;
    default:
        break; // unreachable in this model
    }
}

static void launch_ln(const float* in, const float* w, const float* b, float* out,
                      int ntok, float eps, hipStream_t s)
{
    ln128_kernel<<<cdiv(ntok * 32, 256), 256, 0, s>>>(in, w, b, out, ntok, eps);
}

extern "C" void kernel_launch(void* const* d_in, const int* in_sizes, int n_in,
                              void* d_out, int out_size, void* d_ws, size_t ws_size,
                              hipStream_t stream)
{
    const float* x        = (const float*)d_in[0];
    const float* td_w     = (const float*)d_in[1];
    const float* td_ln_w  = (const float*)d_in[2];
    const float* td_ln_b  = (const float*)d_in[3];
    const float* ln1_w    = (const float*)d_in[4];
    const float* ln1_b    = (const float*)d_in[5];
    const float* m_in_w   = (const float*)d_in[6];
    const float* m_conv_w = (const float*)d_in[7];
    const float* m_conv_b = (const float*)d_in[8];
    const float* m_xproj_w= (const float*)d_in[9];
    const float* m_dt_w   = (const float*)d_in[10];
    const float* m_dt_b   = (const float*)d_in[11];
    const float* m_Alog   = (const float*)d_in[12];
    const float* m_D      = (const float*)d_in[13];
    const float* m_out_w  = (const float*)d_in[14];
    const float* ln2_w    = (const float*)d_in[15];
    const float* ln2_b    = (const float*)d_in[16];
    const float* qkv_w    = (const float*)d_in[17];
    const float* qkv_b    = (const float*)d_in[18];
    const float* aout_w   = (const float*)d_in[19];
    const float* aout_b   = (const float*)d_in[20];
    const float* ln3_w    = (const float*)d_in[21];
    const float* ln3_b    = (const float*)d_in[22];
    const float* fc1_w    = (const float*)d_in[23];
    const float* fc1_b    = (const float*)d_in[24];
    const float* fc2_w    = (const float*)d_in[25];
    const float* fc2_b    = (const float*)d_in[26];
    const float* fnorm_w  = (const float*)d_in[27];
    const float* fnorm_b  = (const float*)d_in[28];
    const float* head_w   = (const float*)d_in[29];
    const float* head_b   = (const float*)d_in[30];
    float* out = (float*)d_out;

    // Workspace layout (floats).  big1/big2/big3 are aliased across phases.
    float* ws   = (float*)d_ws;
    float* xs   = ws;                       // 2,097,152  residual stream
    float* hln  = xs   + 2097152;           // 2,097,152  LN output
    float* big1 = hln  + 2097152;           // 8,388,608  feats | xz | a1
    float* big2 = big1 + 8388608;           // 8,388,608  im2col | xc+delta | qkv
    float* big3 = big2 + 8388608;           // 4,194,304  stem-conv | y | attno
    float* dbc  = big3 + 4194304;           //   655,360
    float* xc    = big2;                    // [NTOK,256]
    float* delta = big2 + 4194304;          // [NTOK,256]
    float* qkv   = big2;                    // [NTOK,384] (after mamba phase)
    float* y     = big3;                    // [NTOK,256]
    float* attno = big3;                    // [NTOK,128] (after mamba phase)
    float* a1    = big1;                    // [NTOK,512] (MLP hidden)

    const int heads[6] = {2, 2, 4, 2, 2, 4};

    // ---- Stem: mean over Hf, stride-4 conv as im2col GEMM, LayerNorm ----
    mean_h_kernel <<<cdiv(8388608, 256), 256, 0, stream>>>(x, big1, 8388608);
    im2col_kernel <<<cdiv(8388608, 256), 256, 0, stream>>>(big1, big2, 8388608);
    launch_gemm(big2, td_w, nullptr, big3, NTOK, 128, 512, GOP_STORE, stream);
    launch_ln(big3, td_ln_w, td_ln_b, xs, NTOK, 1e-5f, stream);

    for (int l = 0; l < 6; ++l) {
        // ---- bidirectional Mamba ----
        launch_ln(xs, ln1_w + l * 128, ln1_b + l * 128, hln, NTOK, 1e-5f, stream);
        for (int dir = 0; dir < 2; ++dir) {
            size_t ld = (size_t)(l * 2 + dir);
            launch_gemm(hln, m_in_w + ld * 512 * 128, nullptr, big1,
                        NTOK, 512, 128, GOP_STORE, stream);
            conv_silu_kernel<<<cdiv(NTOK * 256, 256), 256, 0, stream>>>(
                big1, m_conv_w + ld * 256 * 4, m_conv_b + ld * 256, xc, dir, NTOK * 256);
            launch_gemm(xc, m_xproj_w + ld * 40 * 256, nullptr, dbc,
                        NTOK, 40, 256, GOP_STORE, stream);
            dt_softplus_kernel<<<cdiv(NTOK * 256, 256), 256, 0, stream>>>(
                dbc, m_dt_w + ld * 256 * 8, m_dt_b + ld * 256, delta, NTOK * 256);
            scan_kernel<<<NB, 256, 0, stream>>>(
                delta, dbc, xc, big1, m_Alog + ld * 256 * 16, m_D + ld * 256, y, dir);
            launch_gemm(y, m_out_w + ld * 128 * 256, nullptr, xs,
                        NTOK, 128, 256, GOP_ACCUM, stream);
        }
        // ---- attention ----
        launch_ln(xs, ln2_w + l * 128, ln2_b + l * 128, hln, NTOK, 1e-5f, stream);
        launch_gemm(hln, qkv_w + (size_t)l * 384 * 128, qkv_b + l * 384, qkv,
                    NTOK, 384, 128, GOP_STORE, stream);
        int nh = heads[l];
        int waves = NB * nh * (NT / 16);
        if (nh == 2)
            attn_flash_kernel<64><<<cdiv(waves, 4), 128, 0, stream>>>(qkv, attno, waves);
        else
            attn_flash_kernel<32><<<cdiv(waves, 4), 128, 0, stream>>>(qkv, attno, waves);
        launch_gemm(attno, aout_w + (size_t)l * 128 * 128, aout_b + l * 128, xs,
                    NTOK, 128, 128, GOP_ACCUM, stream);
        // ---- MLP ----
        launch_ln(xs, ln3_w + l * 128, ln3_b + l * 128, hln, NTOK, 1e-5f, stream);
        launch_gemm(hln, fc1_w + (size_t)l * 512 * 128, fc1_b + l * 512, a1,
                    NTOK, 512, 128, GOP_GELU, stream);
        launch_gemm(a1, fc2_w + (size_t)l * 128 * 512, fc2_b + l * 128, xs,
                    NTOK, 128, 512, GOP_ACCUM, stream);
    }

    // ---- final norm + head ----
    launch_ln(xs, fnorm_w, fnorm_b, hln, NTOK, 1e-6f, stream);
    launch_gemm(hln, head_w, head_b, out, NTOK, 90, 128, GOP_STORE, stream);
    (void)in_sizes; (void)n_in; (void)out_size; (void)ws_size;
}
